// Cell_49503793054251
// MI455X (gfx1250) — matrix operand order repaired and checked by reading, not compiled, not run on hardware
//
#include <hip/hip_runtime.h>

// ---------------- problem constants ----------------
#define B_      64
#define N_      256
#define D_      512
#define E_      131072
#define H_      8
#define TOTAL_  16384
#define DH_     64

// ---------------- types ----------------
typedef __bf16 bf16;
typedef __attribute__((ext_vector_type(16))) __bf16 v16bf;
typedef __attribute__((ext_vector_type(8)))  __bf16 bf16x8;
typedef __attribute__((ext_vector_type(8)))  float  v8f;

union BF16V { v16bf v; bf16x8 h[2]; };

__device__ __forceinline__ bf16 to_bf16(float f) {
  unsigned u = __float_as_uint(f);
  unsigned r = u + 0x7fffu + ((u >> 16) & 1u);   // round-to-nearest-even
  unsigned short s = (unsigned short)(r >> 16);
  bf16 b;
  __builtin_memcpy(&b, &s, 2);
  return b;
}

__device__ __forceinline__ v8f zero8() {
  v8f z = {0.f, 0.f, 0.f, 0.f, 0.f, 0.f, 0.f, 0.f};
  return z;
}

// low 32 bits of a generic pointer into LDS == workgroup-relative LDS offset
__device__ __forceinline__ unsigned lds_off_u32(const void* p) {
  return (unsigned)(uintptr_t)p;
}

// CDNA5 async global->LDS copy (ASYNCcnt-tracked, no VGPR round trip)
__device__ __forceinline__ void async_copy_b128(unsigned lds_off, const void* gptr) {
  asm volatile("global_load_async_to_lds_b128 %0, %1, off"
               :
               : "v"(lds_off), "v"((unsigned long long)(uintptr_t)gptr)
               : "memory");
}
__device__ __forceinline__ void wait_async_le4() {
  asm volatile("s_wait_asynccnt 0x4" ::: "memory");
}
__device__ __forceinline__ void wait_async_0() {
  asm volatile("s_wait_asynccnt 0x0" ::: "memory");
}

// ============================================================
// GEMM: C[M x Nout] = act( A[M x K](bf16) @ W[Nout x K]^T + bias ) (+ resid)
// block tile 128x128, 8 waves, each wave 32(M) x 64(N), K-step 32, WMMA bf16.
// Global->LDS staging uses async-to-LDS with double buffering.
// ============================================================
__global__ __launch_bounds__(256) void gemm_bf16_kernel(
    const bf16* __restrict__ A, const bf16* __restrict__ W,
    const float* __restrict__ bias, const float* __restrict__ resid,
    float* __restrict__ outF, bf16* __restrict__ outB,
    int K, int Nout, int relu)
{
  __shared__ __align__(16) bf16 sA[2][128][32];   // 2 x 8 KB
  __shared__ __align__(16) bf16 sB[2][128][32];   // 2 x 8 KB

  const int tid  = threadIdx.x;
  const int wave = tid >> 5;
  const int lane = tid & 31;
  const int lh   = lane & 15;
  const int hi   = lane >> 4;
  const int m0 = blockIdx.y * 128;
  const int n0 = blockIdx.x * 128;
  const int wm = (wave & 3) * 32;   // wave M offset inside tile
  const int wn = (wave >> 2) * 64;  // wave N offset inside tile

  // each thread copies one 32B chunk of A and of B per K-tile (2x b128 each)
  const int lr = tid >> 1;            // tile row 0..127
  const int lc = (tid & 1) * 16;      // element col 0 or 16
  const bf16* gA = &A[(size_t)(m0 + lr) * K + lc];
  const bf16* gB = &W[(size_t)(n0 + lr) * K + lc];
  unsigned ldsA[2], ldsB[2];
  ldsA[0] = lds_off_u32(&sA[0][lr][lc]);
  ldsA[1] = lds_off_u32(&sA[1][lr][lc]);
  ldsB[0] = lds_off_u32(&sB[0][lr][lc]);
  ldsB[1] = lds_off_u32(&sB[1][lr][lc]);

  auto issue_tile = [&](int buf, int k0) {
    async_copy_b128(ldsA[buf],      gA + k0);
    async_copy_b128(ldsA[buf] + 16, gA + k0 + 8);
    async_copy_b128(ldsB[buf],      gB + k0);
    async_copy_b128(ldsB[buf] + 16, gB + k0 + 8);
  };

  v8f acc[2][4];
#pragma unroll
  for (int i = 0; i < 2; ++i)
#pragma unroll
    for (int j = 0; j < 4; ++j) acc[i][j] = zero8();

  const int T = K >> 5;     // K tiles of 32
  issue_tile(0, 0);

  for (int t = 0; t < T; ++t) {
    const int cur = t & 1;
    if (t + 1 < T) {
      issue_tile(cur ^ 1, (t + 1) * 32);  // prefetch next tile into other buffer
      wait_async_le4();                   // in-order: tile t's 4 copies landed
    } else {
      wait_async_0();
    }
    __syncthreads();

    // A fragment (16x32 bf16): lane<16 -> M=lh, K = {0..7,16..23}; lane>=16 -> K+8
    v16bf af[2], bfr[4];
#pragma unroll
    for (int i = 0; i < 2; ++i) {
      BF16V u;
      int mr = wm + i * 16 + lh;
      int kb = hi * 8;
      u.h[0] = *(const bf16x8*)&sA[cur][mr][kb];
      u.h[1] = *(const bf16x8*)&sA[cur][mr][kb + 16];
      af[i] = u.v;
    }
    // B fragment (32x16): lane<16 -> N=lh, K = {0..7,16..23}; lane>=16 -> K+8
#pragma unroll
    for (int j = 0; j < 4; ++j) {
      BF16V u;
      int nr = wn + j * 16 + lh;
      int kb = hi * 8;
      u.h[0] = *(const bf16x8*)&sB[cur][nr][kb];
      u.h[1] = *(const bf16x8*)&sB[cur][nr][kb + 16];
      bfr[j] = u.v;
    }
#pragma unroll
    for (int i = 0; i < 2; ++i)
#pragma unroll
      for (int j = 0; j < 4; ++j)
        acc[i][j] = __builtin_amdgcn_wmma_f32_16x16x32_bf16(
            false, af[i], false, bfr[j], (short)0, acc[i][j], false, false);
    __syncthreads();   // all waves done reading buf[cur] before it is refilled
  }

  // epilogue: C layout -> VGPR r: lanes<16 M=r, lanes>=16 M=r+8; N=lh
#pragma unroll
  for (int i = 0; i < 2; ++i)
#pragma unroll
    for (int j = 0; j < 4; ++j) {
      int col   = n0 + wn + j * 16 + lh;
      int rbase = m0 + wm + i * 16 + hi * 8;
#pragma unroll
      for (int r = 0; r < 8; ++r) {
        int row = rbase + r;
        float v = acc[i][j][r];
        if (bias)  v += bias[col];
        if (relu)  v = fmaxf(v, 0.0f);
        if (resid) v += resid[(size_t)row * Nout + col];
        size_t o = (size_t)row * Nout + col;
        if (outF) outF[o] = v;
        if (outB) outB[o] = to_bf16(v);
      }
    }
}

// ============================================================
// Attention: one block per (b, h, half); 8 waves x 16 query rows.
// qkv: [TOTAL x 1536] bf16 (q|k|v), vt: [B*H*64 x 256] bf16 (V transposed)
// out: [TOTAL x 512] bf16
// ============================================================
__global__ __launch_bounds__(256) void attn_kernel(
    const bf16* __restrict__ qkv, const bf16* __restrict__ vt,
    bf16* __restrict__ out)
{
  __shared__ __align__(16) bf16 sp[8][16 * 256];   // 64 KB, wave-private P tiles

  const int tid  = threadIdx.x;
  const int wave = tid >> 5;
  const int lane = tid & 31;
  const int lh   = lane & 15;
  const int hi   = lane >> 4;
  const int half = blockIdx.x & 1;
  const int bh   = blockIdx.x >> 1;
  const int b    = bh >> 3;
  const int h    = bh & 7;
  const int q0   = half * 128 + wave * 16;

  // Q fragments: 16 rows x 64 d -> 2 (16x32) fragments
  v16bf qa[2];
  {
    const int s = q0 + lh;
    const size_t base = ((size_t)(b * N_ + s)) * 1536 + h * 64;
#pragma unroll
    for (int c = 0; c < 2; ++c) {
      BF16V u;
      int d0 = c * 32 + hi * 8;
      u.h[0] = *(const bf16x8*)&qkv[base + d0];
      u.h[1] = *(const bf16x8*)&qkv[base + d0 + 16];
      qa[c] = u.v;
    }
  }

  // scores for all 256 keys: 16 accumulators of 16x16
  v8f sacc[16];
#pragma unroll
  for (int kt = 0; kt < 16; ++kt) {
    const int t = kt * 16 + lh;
    const size_t kb = ((size_t)(b * N_ + t)) * 1536 + 512 + h * 64;
    v8f z = zero8();
#pragma unroll
    for (int c = 0; c < 2; ++c) {
      BF16V u;
      int d0 = c * 32 + hi * 8;
      u.h[0] = *(const bf16x8*)&qkv[kb + d0];
      u.h[1] = *(const bf16x8*)&qkv[kb + d0 + 16];
      z = __builtin_amdgcn_wmma_f32_16x16x32_bf16(false, qa[c], false, u.v,
                                                  (short)0, z, false, false);
    }
    sacc[kt] = z;
  }

  // softmax over the 256-key rows (scale = 1/sqrt(64) = 0.125)
#pragma unroll
  for (int r = 0; r < 8; ++r) {
    float m = -3.0e38f;
#pragma unroll
    for (int kt = 0; kt < 16; ++kt) m = fmaxf(m, sacc[kt][r]);
#pragma unroll
    for (int o = 1; o < 16; o <<= 1) m = fmaxf(m, __shfl_xor(m, o, 32));
    float ssum = 0.f;
#pragma unroll
    for (int kt = 0; kt < 16; ++kt) {
      float p = __expf(0.125f * (sacc[kt][r] - m));
      sacc[kt][r] = p;
      ssum += p;
    }
#pragma unroll
    for (int o = 1; o < 16; o <<= 1) ssum += __shfl_xor(ssum, o, 32);
    float inv = 1.0f / ssum;
#pragma unroll
    for (int kt = 0; kt < 16; ++kt) sacc[kt][r] *= inv;
  }

  // stage P as bf16 in wave-private LDS (row-major 16 x 256)
  bf16* pw = sp[wave];
#pragma unroll
  for (int kt = 0; kt < 16; ++kt)
#pragma unroll
    for (int r = 0; r < 8; ++r)
      pw[(r + hi * 8) * 256 + kt * 16 + lh] = to_bf16(sacc[kt][r]);

  // O = P @ V : K = 256 (8 chained WMMAs), N = 64 (4 fragments)
  v8f oacc[4];
#pragma unroll
  for (int j = 0; j < 4; ++j) oacc[j] = zero8();
#pragma unroll
  for (int kc = 0; kc < 8; ++kc) {
    BF16V ua;
    int kb0 = kc * 32 + hi * 8;
    ua.h[0] = *(const bf16x8*)&pw[lh * 256 + kb0];
    ua.h[1] = *(const bf16x8*)&pw[lh * 256 + kb0 + 16];
#pragma unroll
    for (int j = 0; j < 4; ++j) {
      BF16V ub;
      size_t vb = ((size_t)bh * 64 + j * 16 + lh) * 256 + kc * 32 + hi * 8;
      ub.h[0] = *(const bf16x8*)&vt[vb];
      ub.h[1] = *(const bf16x8*)&vt[vb + 16];
      oacc[j] = __builtin_amdgcn_wmma_f32_16x16x32_bf16(false, ua.v, false, ub.v,
                                                        (short)0, oacc[j], false, false);
    }
  }

#pragma unroll
  for (int j = 0; j < 4; ++j)
#pragma unroll
    for (int r = 0; r < 8; ++r) {
      int s   = q0 + r + hi * 8;
      int col = h * 64 + j * 16 + lh;
      out[((size_t)(b * N_ + s)) * 512 + col] = to_bf16(oacc[j][r]);
    }
}

// ============================================================
// small elementwise / reduction kernels
// ============================================================
__global__ void cvt_bf16_kernel(const float* __restrict__ x, bf16* __restrict__ y,
                                int n, int relu) {
  int i = blockIdx.x * blockDim.x + threadIdx.x;
  if (i >= n) return;
  float v = x[i];
  if (relu) v = fmaxf(v, 0.0f);
  y[i] = to_bf16(v);
}

__global__ void fill_kernel(float* __restrict__ p, float v, int n) {
  int i = blockIdx.x * blockDim.x + threadIdx.x;
  if (i < n) p[i] = v;
}

__global__ void deg_count_kernel(const int* __restrict__ ei, float* __restrict__ deg, int e_cnt) {
  int e = blockIdx.x * blockDim.x + threadIdx.x;
  if (e < e_cnt) atomicAdd(&deg[ei[e]], 1.0f);   // row = edge_index[0][e]
}

__global__ void rsqrt_kernel(const float* __restrict__ x, float* __restrict__ y, int n) {
  int i = blockIdx.x * blockDim.x + threadIdx.x;
  if (i < n) y[i] = rsqrtf(x[i]);
}

// per-edge message + scatter-add: 1 block / edge, 128 threads x float4
__global__ __launch_bounds__(128) void gcn_scatter_kernel(
    const float* __restrict__ xw, const float* __restrict__ ea,
    const int* __restrict__ ei, const float* __restrict__ dinv,
    float* __restrict__ agg, int e_cnt)
{
  int e = blockIdx.x;
  int row = ei[e];
  int col = ei[e_cnt + e];
  float nrm = dinv[row] * dinv[col];
  const float4* xr = (const float4*)(xw + (size_t)row * D_);
  const float4* er = (const float4*)(ea + (size_t)e * D_);
  float* ag = agg + (size_t)col * D_;
  int c = threadIdx.x;
  float4 a = xr[c];
  float4 bv = er[c];
  atomicAdd(&ag[c * 4 + 0], fmaxf(a.x + bv.x, 0.f) * nrm);
  atomicAdd(&ag[c * 4 + 1], fmaxf(a.y + bv.y, 0.f) * nrm);
  atomicAdd(&ag[c * 4 + 2], fmaxf(a.z + bv.z, 0.f) * nrm);
  atomicAdd(&ag[c * 4 + 3], fmaxf(a.w + bv.w, 0.f) * nrm);
}

__global__ void gcn_combine_kernel(const float* __restrict__ agg,
                                   const float* __restrict__ xw,
                                   const float* __restrict__ root,
                                   const float* __restrict__ deg,
                                   float* __restrict__ out) {
  int i = blockIdx.x * blockDim.x + threadIdx.x;
  if (i >= TOTAL_ * D_) return;
  int row = i >> 9, col = i & 511;
  out[i] = agg[i] + fmaxf(xw[i] + root[col], 0.f) / deg[row];
}

// BatchNorm training-mode stats: 1 block per channel
__global__ __launch_bounds__(256) void bn_stats_kernel(const float* __restrict__ x,
                                                       float* __restrict__ mean,
                                                       float* __restrict__ var) {
  __shared__ float r1[256], r2[256];
  int col = blockIdx.x, tid = threadIdx.x;
  float s = 0.f, ss = 0.f;
  for (int r = tid; r < TOTAL_; r += 256) {
    float v = x[(size_t)r * D_ + col];
    s += v; ss += v * v;
  }
  r1[tid] = s; r2[tid] = ss;
  __syncthreads();
  for (int o = 128; o > 0; o >>= 1) {
    if (tid < o) { r1[tid] += r1[tid + o]; r2[tid] += r2[tid + o]; }
    __syncthreads();
  }
  if (tid == 0) {
    float m = r1[0] * (1.0f / TOTAL_);
    mean[col] = m;
    var[col] = r2[0] * (1.0f / TOTAL_) - m * m;   // biased var
  }
}

__global__ void bn_apply_kernel(float* __restrict__ x, const float* __restrict__ g,
                                const float* __restrict__ be, const float* __restrict__ mean,
                                const float* __restrict__ var) {
  int i = blockIdx.x * blockDim.x + threadIdx.x;
  if (i >= TOTAL_ * D_) return;
  int col = i & 511;
  x[i] = g[col] * (x[i] - mean[col]) * rsqrtf(var[col] + 1e-5f) + be[col];
}

// LayerNorm per row -> bf16 output
__global__ __launch_bounds__(256) void ln_kernel(const float* __restrict__ x,
                                                 const float* __restrict__ g,
                                                 const float* __restrict__ be,
                                                 bf16* __restrict__ out) {
  __shared__ float red[256];
  int row = blockIdx.x, tid = threadIdx.x;
  const float* xr = x + (size_t)row * D_;
  float a0 = xr[tid], a1 = xr[tid + 256];
  red[tid] = a0 + a1;
  __syncthreads();
  for (int o = 128; o > 0; o >>= 1) { if (tid < o) red[tid] += red[tid + o]; __syncthreads(); }
  float mean = red[0] * (1.0f / D_);
  __syncthreads();
  float d0 = a0 - mean, d1 = a1 - mean;
  red[tid] = d0 * d0 + d1 * d1;
  __syncthreads();
  for (int o = 128; o > 0; o >>= 1) { if (tid < o) red[tid] += red[tid + o]; __syncthreads(); }
  float inv = rsqrtf(red[0] * (1.0f / D_) + 1e-5f);
  size_t o = (size_t)row * D_;
  out[o + tid]       = to_bf16(g[tid] * d0 * inv + be[tid]);
  out[o + tid + 256] = to_bf16(g[tid + 256] * d1 * inv + be[tid + 256]);
}

// zero last node of each graph (the module's off-by-one pad behavior)
__global__ void mask_copy_kernel(const float* __restrict__ x, float* __restrict__ y) {
  int i = blockIdx.x * blockDim.x + threadIdx.x;
  if (i >= TOTAL_ * D_) return;
  int row = i >> 9;
  y[i] = ((row & (N_ - 1)) == N_ - 1) ? 0.0f : x[i];
}

// transpose V into [B*H*DH, N] for contiguous WMMA B-fragment loads
__global__ void vtrans_kernel(const bf16* __restrict__ qkv, bf16* __restrict__ vt) {
  int i = blockIdx.x * blockDim.x + threadIdx.x;
  if (i >= TOTAL_ * D_) return;
  int t  = i & 255;
  int dh = (i >> 8) & 63;
  int bh = i >> 14;
  int b = bh >> 3, h = bh & 7;
  vt[i] = qkv[((size_t)(b * N_ + t)) * 1536 + 1024 + h * 64 + dh];
}

// m = gcn + mask(y); also write strided into d_out
__global__ void final_combine_kernel(const float* __restrict__ g, const float* __restrict__ y,
                                     float* __restrict__ m, float* __restrict__ out, int coloff) {
  int i = blockIdx.x * blockDim.x + threadIdx.x;
  if (i >= TOTAL_ * D_) return;
  int row = i >> 9, col = i & 511;
  float v = g[i] + (((row & (N_ - 1)) == N_ - 1) ? 0.0f : y[i]);
  if (m) m[i] = v;
  out[(size_t)row * 1024 + coloff + col] = v;
}

// ============================================================
// host-side orchestration
// ============================================================
extern "C" void kernel_launch(void* const* d_in, const int* in_sizes, int n_in,
                              void* d_out, int out_size, void* d_ws, size_t ws_size,
                              hipStream_t stream) {
  (void)in_sizes; (void)n_in; (void)out_size; (void)ws_size;

  // flattened input order (dict DFS order from setup_inputs):
  // 0 s0, 1 s1, 2 edge_attr,
  // 3 pre0.w 4 pre0.b 5 pre0.g 6 pre0.be, 7 pre1.w 8 pre1.b 9 pre1.g 10 pre1.be,
  // 11 gcn0.w 12 gcn0.b 13 gcn0.root 14 gcn0.g 15 gcn0.be,
  // 16 gcn1.w 17 gcn1.b 18 gcn1.root 19 gcn1.g 20 gcn1.be,
  // 21..32 tr0: w_in b_in w_out b_out w1 b1 w2 b2 ln1g ln1be ln2g ln2be,
  // 33..44 tr1: same, 45 edge_index, 46 batch, 47..50 unused
  const float* s0        = (const float*)d_in[0];
  const float* s1        = (const float*)d_in[1];
  const float* edge_attr = (const float*)d_in[2];
  const int*   edge_idx  = (const int*)d_in[45];
  float* outp = (float*)d_out;

  // ---- carve workspace ----
  char* ws = (char*)d_ws;
  size_t off = 0;
  auto carve = [&](size_t bytes) -> void* {
    void* p = ws + off;
    off += (bytes + 255) & ~(size_t)255;
    return p;
  };

  // bf16 weights: pre0, pre1, gcn0, gcn1, tr0{win,wout,w1,w2}, tr1{...}
  const int    widx[12] = {3, 7, 11, 16, 21, 23, 25, 27, 33, 35, 37, 39};
  const size_t wsz[12]  = {262144, 262144, 262144, 262144,
                           786432, 262144, 524288, 524288,
                           786432, 262144, 524288, 524288};
  bf16* wbf[12];
  for (int i = 0; i < 12; ++i) wbf[i] = (bf16*)carve(wsz[i] * 2);

  bf16* act_bf = (bf16*)carve((size_t)TOTAL_ * 512 * 2);
  bf16* hbf    = (bf16*)carve((size_t)TOTAL_ * 1024 * 2);
  bf16* kqv_bf = (bf16*)carve((size_t)TOTAL_ * 1536 * 2);
  bf16* vt_bf  = (bf16*)carve((size_t)TOTAL_ * 512 * 2);

  float* n0    = (float*)carve((size_t)TOTAL_ * 512 * 4);
  float* n1    = (float*)carve((size_t)TOTAL_ * 512 * 4);
  float* m2b   = (float*)carve((size_t)TOTAL_ * 512 * 4);
  float* ebuf  = (float*)carve((size_t)TOTAL_ * 512 * 4);
  float* x1    = (float*)carve((size_t)TOTAL_ * 512 * 4);
  float* ybuf  = (float*)carve((size_t)TOTAL_ * 512 * 4);
  float* gout  = (float*)carve((size_t)TOTAL_ * 512 * 4);
  float* aggb  = (float*)carve((size_t)TOTAL_ * 512 * 4);
  float* xw    = (float*)carve((size_t)TOTAL_ * 512 * 4);
  float* degb  = (float*)carve((size_t)TOTAL_ * 4);
  float* dinv  = (float*)carve((size_t)TOTAL_ * 4);
  float* meanb = (float*)carve(512 * 4);
  float* varb  = (float*)carve(512 * 4);

  const int ELT = TOTAL_ * 512;                 // 8 388 608
  const dim3 G_ELT((ELT + 255) / 256), T256(256);

  // ---- weight conversion (deterministic, every call) ----
  for (int i = 0; i < 12; ++i) {
    int n = (int)wsz[i];
    cvt_bf16_kernel<<<dim3((n + 255) / 256), T256, 0, stream>>>(
        (const float*)d_in[widx[i]], wbf[i], n, 0);
  }

  // ---- degrees: deg = 1 + count(row); dinv = deg^-0.5 ----
  fill_kernel<<<dim3(TOTAL_ / 256), T256, 0, stream>>>(degb, 1.0f, TOTAL_);
  deg_count_kernel<<<dim3(E_ / 256), T256, 0, stream>>>(edge_idx, degb, E_);
  rsqrt_kernel<<<dim3(TOTAL_ / 256), T256, 0, stream>>>(degb, dinv, TOTAL_);

  auto gemm = [&](const bf16* A, const bf16* W, const float* bias, const float* resid,
                  float* oF, bf16* oB, int K, int Nout, int relu) {
    gemm_bf16_kernel<<<dim3(Nout / 128, TOTAL_ / 128), T256, 0, stream>>>(
        A, W, bias, resid, oF, oB, K, Nout, relu);
  };
  auto bn = [&](float* x, const float* g, const float* be) {
    bn_stats_kernel<<<dim3(512), T256, 0, stream>>>(x, meanb, varb);
    bn_apply_kernel<<<G_ELT, T256, 0, stream>>>(x, g, be, meanb, varb);
  };

  // ---- pre layers: n = BN(Linear(s)) ----
  cvt_bf16_kernel<<<G_ELT, T256, 0, stream>>>(s0, act_bf, ELT, 0);
  gemm(act_bf, wbf[0], (const float*)d_in[4], nullptr, n0, nullptr, 512, 512, 0);
  bn(n0, (const float*)d_in[5], (const float*)d_in[6]);

  cvt_bf16_kernel<<<G_ELT, T256, 0, stream>>>(s1, act_bf, ELT, 0);
  gemm(act_bf, wbf[1], (const float*)d_in[8], nullptr, n1, nullptr, 512, 512, 0);
  bn(n1, (const float*)d_in[9], (const float*)d_in[10]);

  // ---- GCN branch ----
  auto run_gcn = [&](const float* xin, int wi, int bi, int ri, int gi, int bei, float* outb) {
    cvt_bf16_kernel<<<G_ELT, T256, 0, stream>>>(xin, act_bf, ELT, 1);   // relu(x)
    gemm(act_bf, wbf[wi], (const float*)d_in[bi], nullptr, xw, nullptr, 512, 512, 0);
    fill_kernel<<<G_ELT, T256, 0, stream>>>(aggb, 0.0f, ELT);
    gcn_scatter_kernel<<<dim3(E_), dim3(128), 0, stream>>>(
        xw, edge_attr, edge_idx, dinv, aggb, E_);
    gcn_combine_kernel<<<G_ELT, T256, 0, stream>>>(
        aggb, xw, (const float*)d_in[ri], degb, outb);
    bn(outb, (const float*)d_in[gi], (const float*)d_in[bei]);
  };

  // ---- transformer branch (norm_first, relu FFN) ----
  auto run_tr = [&](const float* xin, int base, int wb, float* yout) {
    mask_copy_kernel<<<G_ELT, T256, 0, stream>>>(xin, ebuf);
    ln_kernel<<<dim3(TOTAL_), T256, 0, stream>>>(
        ebuf, (const float*)d_in[base + 8], (const float*)d_in[base + 9], act_bf);
    // qkv projection (bf16 out only)
    gemm(act_bf, wbf[wb + 0], (const float*)d_in[base + 1], nullptr,
         nullptr, kqv_bf, 512, 1536, 0);
    vtrans_kernel<<<G_ELT, T256, 0, stream>>>(kqv_bf, vt_bf);
    attn_kernel<<<dim3(B_ * H_ * 2), T256, 0, stream>>>(kqv_bf, vt_bf, act_bf);
    // out projection + residual with e
    gemm(act_bf, wbf[wb + 1], (const float*)d_in[base + 3], ebuf,
         x1, nullptr, 512, 512, 0);
    ln_kernel<<<dim3(TOTAL_), T256, 0, stream>>>(
        x1, (const float*)d_in[base + 10], (const float*)d_in[base + 11], act_bf);
    // FFN up (relu, bf16 out), FFN down (+residual x1)
    gemm(act_bf, wbf[wb + 2], (const float*)d_in[base + 5], nullptr,
         nullptr, hbf, 512, 1024, 1);
    gemm(hbf, wbf[wb + 3], (const float*)d_in[base + 7], x1,
         yout, nullptr, 1024, 512, 0);
  };

  // ---- m2 = GCN0(n0) + PadTransform0(n1) ----
  run_gcn(n0, 2, 12, 13, 14, 15, gout);
  run_tr(n1, 21, 4, ybuf);
  final_combine_kernel<<<G_ELT, T256, 0, stream>>>(gout, ybuf, m2b, outp, 0);

  // ---- m3 = GCN1(n1) + PadTransform1(m2) ----
  run_gcn(n1, 3, 17, 18, 19, 20, gout);
  run_tr(m2b, 33, 8, ybuf);
  final_combine_kernel<<<G_ELT, T256, 0, stream>>>(gout, ybuf, nullptr, outp, 512);
}
